// ConvSelfAttention_52132313039213
// MI455X (gfx1250) — matrix operand order repaired and checked by reading, not compiled, not run on hardware
//
#include <hip/hip_runtime.h>
#include <hip/hip_bf16.h>
#include <cstddef>

typedef _Float16 f16;
typedef __attribute__((ext_vector_type(16))) _Float16 v16h;
typedef __attribute__((ext_vector_type(8)))  _Float16 v8h;
typedef __attribute__((ext_vector_type(8)))  float    v8f;
typedef __attribute__((ext_vector_type(4)))  unsigned int u32x4;
typedef __attribute__((ext_vector_type(8)))  int      i32x8;
typedef __attribute__((ext_vector_type(4)))  int      i32x4;

#define EPSF 1e-9f

// Tensor Data Mover path. Guarded so the file compiles on toolchains without
// the builtin (falls back to global-load K fragments). Arity differs between
// toolchains: clang-22/ROCm7.2 = 5 args, clang-23/therock = 6 args.
#if defined(__has_builtin)
#if __has_builtin(__builtin_amdgcn_tensor_load_to_lds) && \
    __has_builtin(__builtin_amdgcn_s_wait_tensorcnt)
#define USE_TDM 1
#endif
#endif
#ifndef USE_TDM
#define USE_TDM 0
#endif

// Attention LDS layout (bytes): P | Vt | f32 area | Klds (TDM staging)
#define ATT_P_OFF    0
#define ATT_VT_OFF   32768
#define ATT_F32_OFF  (32768 + 65536)
#define ATT_KLDS_OFF 101632
#define ATT_SMEM     (ATT_KLDS_OFF + 131072)   // 232 KB < 320 KB/WGP

// ---------------------------------------------------------------------------
// A-fragment loader: 16-bit A matrix 16x32, this code's contiguous k order (B loads match):
//   lanes 0-15 : row M=lane,    halves 0..15 = K[0..15]
//   lanes 16-31: row M=lane-16, halves 0..15 = K[16..31]
// Caller passes p = rowBase + kBase + off0 where off0 = (lane<16 ? 0 : 16).
// ---------------------------------------------------------------------------
__device__ __forceinline__ v16h load_a16(const f16* p) {
    v8h lo = *(const v8h*)p;
    v8h hi = *(const v8h*)(p + 8);
    v16h r;
#pragma unroll
    for (int i = 0; i < 8; ++i) { r[i] = lo[i]; r[i + 8] = hi[i]; }
    return r;
}

// ---------------------------------------------------------------------------
// Per-batch mean / rstd over C*H*W = 524288 elements (grid = B blocks)
// ---------------------------------------------------------------------------
__global__ __launch_bounds__(256) void gn_stats_kernel(const float* __restrict__ x,
                                                       float* __restrict__ stats) {
    const int b = blockIdx.x;
    const float* p = x + (size_t)b * 512 * 1024;
    float s = 0.f, sq = 0.f;
    for (int i = threadIdx.x; i < 524288; i += 256) {
        float v = p[i]; s += v; sq += v * v;
    }
#pragma unroll
    for (int m = 16; m >= 1; m >>= 1) { s += __shfl_xor(s, m, 32); sq += __shfl_xor(sq, m, 32); }
    __shared__ float rs[8], rq[8];
    if ((threadIdx.x & 31) == 0) { rs[threadIdx.x >> 5] = s; rq[threadIdx.x >> 5] = sq; }
    __syncthreads();
    if (threadIdx.x == 0) {
        s = 0.f; sq = 0.f;
        for (int w = 0; w < 8; ++w) { s += rs[w]; sq += rq[w]; }
        float mu  = s  * (1.0f / 524288.0f);
        float var = sq * (1.0f / 524288.0f) - mu * mu;
        stats[b * 2 + 0] = mu;
        stats[b * 2 + 1] = rsqrtf(var + EPSF);
    }
}

// ---------------------------------------------------------------------------
// Normalize + scale + convert to f16, write zero-padded (34x34) activation
// ---------------------------------------------------------------------------
__global__ __launch_bounds__(256) void pack_xpad_kernel(const float* __restrict__ x,
                                                        const float* __restrict__ gnw,
                                                        const float* __restrict__ stats,
                                                        f16* __restrict__ xpad) {
    int i = blockIdx.x * 256 + threadIdx.x;
    if (i >= 4 * 512 * 34 * 34) return;
    int xx = i % 34; int t = i / 34;
    int yy = t % 34; t /= 34;
    int c  = t & 511;
    int b  = t >> 9;
    f16 v = (f16)0.f;
    if (xx >= 1 && xx <= 32 && yy >= 1 && yy <= 32) {
        float mu = stats[b * 2], rstd = stats[b * 2 + 1];
        float val = x[(((size_t)b * 512 + c) * 32 + (yy - 1)) * 32 + (xx - 1)];
        v = (f16)((val - mu) * rstd * gnw[c]);
    }
    xpad[i] = v;
}

__global__ __launch_bounds__(256) void f32_to_f16_kernel(const float* __restrict__ src,
                                                         f16* __restrict__ dst, int n) {
    int i = blockIdx.x * 256 + threadIdx.x;
    if (i < n) dst[i] = (f16)src[i];
}

// ---------------------------------------------------------------------------
// Implicit-GEMM 3x3 conv (pad 1) via WMMA f16 -> f32, double-buffered LDS.
//   A = W[OC][4608]  (K = ic*9 + kh*3 + kw, native OIHW flattening)
//   B = im2col of padded activation, staged per 32-K chunk into LDS
// Block = 256 threads (8 waves): tile M=128 (16/wave), N=32 (one image row).
// mode 0: write f16 tokens tok[b][pix][OC]; mode 1: f32 NCHW + residual.
// ---------------------------------------------------------------------------
__global__ __launch_bounds__(256) void conv_wmma_kernel(const f16* __restrict__ W,
                                                        const f16* __restrict__ Xp,
                                                        f16* __restrict__ tok_dst,
                                                        float* __restrict__ out_dst,
                                                        const float* __restrict__ res,
                                                        int OC, int mode) {
    __shared__ __align__(64) f16 Bt[2][32][32];   // [buf][pixel-col][k-in-chunk]
    const int tid  = threadIdx.x;
    const int wave = tid >> 5;
    const int lane = tid & 31;
    const int l15  = lane & 15;
    const int hicl = (lane >= 16);
    const int b    = blockIdx.z;
    const int y    = blockIdx.y;                  // output pixel row
    const int mbase = blockIdx.x * 128 + wave * 16;
    const int off0 = hicl ? 16 : 0;
    const int kh2  = hicl ? 16 : 0;
    const int mrow = mbase + l15;
    const f16* xb = Xp + (size_t)b * 512 * 34 * 34;
    const f16* wrow = W + (size_t)mrow * 4608;

    const int colS = tid & 31;   // pixel column this thread stages
    const int kk0  = tid >> 5;   // first k-in-chunk this thread stages

    auto stage = [&](int kc) {
        f16 (*buf)[32] = Bt[kc & 1];
        const int kb = kc * 32;
#pragma unroll
        for (int i = 0; i < 4; ++i) {
            int kk  = kk0 + i * 8;
            int k   = kb + kk;
            int ic  = k / 9;
            int r9  = k - ic * 9;
            int khh = r9 / 3;
            int kww = r9 - khh * 3;
            buf[colS][kk] = xb[((size_t)ic * 34 + (y + khh)) * 34 + (colS + kww)];
        }
    };

    v8f c0 = {}, c1 = {};
    stage(0);
    for (int kc = 0; kc < 144; ++kc) {
        __syncthreads();                 // stage(kc) visible; buf[(kc+1)&1] free
        if (kc + 1 < 144) stage(kc + 1); // overlaps with WMMAs below
        const f16 (*buf)[32] = Bt[kc & 1];
        v16h a = load_a16(wrow + kc * 32 + off0);
        if (kc + 2 < 144) __builtin_prefetch(wrow + kc * 32 + 64, 0, 0);
        v16h b0 = *(const v16h*)&buf[l15][kh2];
        v16h b1 = *(const v16h*)&buf[l15 + 16][kh2];
        c0 = __builtin_amdgcn_wmma_f32_16x16x32_f16(false, a, false, b0, (short)0, c0, false, false);
        c1 = __builtin_amdgcn_wmma_f32_16x16x32_f16(false, a, false, b1, (short)0, c1, false, false);
    }

    const int pixb = y * 32;
    if (mode == 0) {
        const int ocb = mbase + (hicl ? 8 : 0);
#pragma unroll
        for (int nt = 0; nt < 2; ++nt) {
            v8f cc = nt ? c1 : c0;
            int pix = pixb + nt * 16 + l15;
            v8h o;
#pragma unroll
            for (int r = 0; r < 8; ++r) o[r] = (f16)cc[r];
            *(v8h*)&tok_dst[((size_t)(b * 1024 + pix)) * OC + ocb] = o;
        }
    } else {
#pragma unroll
        for (int nt = 0; nt < 2; ++nt) {
            v8f cc = nt ? c1 : c0;
            int pix = pixb + nt * 16 + l15;
#pragma unroll
            for (int r = 0; r < 8; ++r) {
                int oc = mbase + (hicl ? 8 : 0) + r;
                size_t off = ((size_t)(b * 512 + oc)) * 1024 + pix;
                out_dst[off] = cc[r] + res[off];
            }
        }
    }
}

// ---------------------------------------------------------------------------
// RoPE in-place on f16 tokens [b][pos][nheads*64]; one thread per (re,im) pair
// ---------------------------------------------------------------------------
__global__ __launch_bounds__(256) void rope_kernel(f16* __restrict__ tok, int nheads, int total) {
    int p = blockIdx.x * 256 + threadIdx.x;
    if (p >= total) return;
    int i    = p & 31;
    int rest = p >> 5;
    int pos  = (rest / nheads) & 1023;
    size_t base = (size_t)rest * 64 + 2 * i;
    float theta = __expf(-(float)(2 * i) * (9.2103403719761836f / 64.0f)); // 1/10000^(2i/64)
    float ang = (float)pos * theta;
    float s, c;
    __sincosf(ang, &s, &c);
    float re = (float)tok[base];
    float im = (float)tok[base + 1];
    tok[base]     = (f16)(re * c - im * s);
    tok[base + 1] = (f16)(re * s + im * c);
}

// ---------------------------------------------------------------------------
// Differential attention: one block per (qtile16, head, batch); 8 waves.
//   K panel bulk-copied to LDS by the Tensor Data Mover (when available);
//   scores (branches m=0/1) via WMMA; 2-pass softmax with wave32 width-16
//   shuffles + LDS cross-wave reduce; combine with lambda; P@V via WMMA with
//   ds_add_f32 accumulation; RMS-norm epilogue.
// ---------------------------------------------------------------------------
__global__ __launch_bounds__(256) void diffattn_kernel(const f16* __restrict__ tokq,
                                                       const f16* __restrict__ tokk,
                                                       const f16* __restrict__ tokv,
                                                       const float* __restrict__ lq1,
                                                       const float* __restrict__ lq2,
                                                       const float* __restrict__ lk1,
                                                       const float* __restrict__ lk2,
                                                       const float* __restrict__ linit,
                                                       const float* __restrict__ rmsw,
                                                       f16* __restrict__ aout) {
    extern __shared__ __align__(128) char smem[];
    f16*   P       = (f16*)(smem + ATT_P_OFF);    // 16x1024 f16
    f16*   Vt      = (f16*)(smem + ATT_VT_OFF);   // 32x1024 f16 (transposed V)
    float* O       = (float*)(smem + ATT_F32_OFF);// 16x32 f32
    float* wred    = O + 512;                     // [2][8][16]
    float* rowstat = wred + 256;                  // [2][16] (max, then sum)
    float* lamsh   = rowstat + 32;                // {lambda, 1-lam_init}
    float* rmssh   = lamsh + 2;                   // [16]

    const int tid  = threadIdx.x;
    const int wave = tid >> 5;
    const int lane = tid & 31;
    const int l15  = lane & 15;
    const int hicl = (lane >= 16);
    const int qt   = blockIdx.x;
    const int h    = blockIdx.y;
    const int b    = blockIdx.z;
    const int kvh  = h >> 2;
    const int off0 = hicl ? 16 : 0;
    const int kh2  = hicl ? 16 : 0;
    const int rb   = hicl ? 8 : 0;

    for (int i = tid; i < 512; i += 256) O[i] = 0.f;
    if (wave == 0) {
        float p1 = lq1[h * 32 + lane] * lk1[h * 32 + lane];
        float p2 = lq2[h * 32 + lane] * lk2[h * 32 + lane];
#pragma unroll
        for (int m = 16; m >= 1; m >>= 1) { p1 += __shfl_xor(p1, m, 32); p2 += __shfl_xor(p2, m, 32); }
        if (lane == 0) {
            float li = linit[h];
            lamsh[0] = -(__expf(p1) - __expf(p2) + li);  // lam * (-1)^1
            lamsh[1] = 1.0f - li;
        }
    }

#if USE_TDM
    // TDM bulk copy of the K panel [1024 rows][64 f16] (row stride 256 f16)
    // into Klds. D# per CDNA5 ISA 8.3/8.4: count=1, type=2 (image),
    // data_size=2B, tile 64x1024, tensor_dim0_stride=256.
    f16* Klds = (f16*)(smem + ATT_KLDS_OFF);
    if (wave == 0) {
        typedef __attribute__((address_space(3))) char lds_char_t;
        unsigned ldsoff = (unsigned)(size_t)(lds_char_t*)Klds;
        unsigned long long ga =
            (unsigned long long)(const char*)(tokk + ((size_t)b * 1024) * 256 + kvh * 64);
        u32x4 g0 = { 1u,                                    // count=1, user mode
                     ldsoff,                                // lds_addr (bytes)
                     (unsigned)(ga & 0xffffffffu),          // global_addr[31:0]
                     (unsigned)((ga >> 32) & 0x1ffffffu) | (2u << 30) }; // addr[56:32] | type=2
        i32x8 g1 = { 0x00010000,     // workgroup_mask=0, data_size=1 (2 bytes)
                     64 << 16,       // tensor_dim0[15:0]=64
                     1024 << 16,     // tensor_dim0[31:16]=0, tensor_dim1[15:0]=1024
                     64 << 16,       // tensor_dim1[31:16]=0, tile_dim0=64
                     1024,           // tile_dim1=1024, tile_dim2=0
                     256,            // tensor_dim0_stride[31:0]=256
                     0, 0 };         // stride hi, tensor_dim1_stride=0
        i32x4 gz4 = { 0, 0, 0, 0 };
#if __clang_major__ >= 23
        i32x8 gz8 = { 0, 0, 0, 0, 0, 0, 0, 0 };
        __builtin_amdgcn_tensor_load_to_lds(g0, g1, gz4, gz4, gz8, 0);
#else
        __builtin_amdgcn_tensor_load_to_lds(g0, g1, gz4, gz4, 0);
#endif
        __builtin_amdgcn_s_wait_tensorcnt(0);
    }
#endif

    // stage V transposed: Vt[hd][kpos]
    const f16* vsrc = tokv + (size_t)b * 1024 * 128 + kvh * 32;
    for (int i = tid; i < 32768; i += 256) {
        int hd = i >> 10, kp = i & 1023;
        Vt[hd * 1024 + kp] = vsrc[(size_t)kp * 128 + hd];
    }
    __syncthreads();

    // ---------------- phase 1: scores for both branches -----------------
    const int qrow = qt * 16 + l15;
    const f16* qbase = tokq + ((size_t)(b * 1024 + qrow)) * 1024 + h * 64;
    v16h qf0 = load_a16(qbase + 0  + off0);
    v16h qf1 = load_a16(qbase + 32 + off0);
    v8f s0[8], s1[8];
#pragma unroll
    for (int j = 0; j < 8; ++j) {
        int kpos = (wave + j * 8) * 16 + l15;
#if USE_TDM
        const f16* kbase = Klds + (size_t)kpos * 64;
#else
        const f16* kbase = tokk + ((size_t)(b * 1024 + kpos)) * 256 + kvh * 64;
#endif
        v16h k0 = *(const v16h*)(kbase + 0  + kh2);
        v16h k1 = *(const v16h*)(kbase + 32 + kh2);
        v8f z0 = {}, z1 = {};
        s0[j] = __builtin_amdgcn_wmma_f32_16x16x32_f16(false, qf0, false, k0, (short)0, z0, false, false);
        s1[j] = __builtin_amdgcn_wmma_f32_16x16x32_f16(false, qf1, false, k1, (short)0, z1, false, false);
    }
    const float sc = 0.17677669529663689f;  // 1/sqrt(32)
#pragma unroll
    for (int j = 0; j < 8; ++j)
#pragma unroll
        for (int r = 0; r < 8; ++r) { s0[j][r] *= sc; s1[j][r] *= sc; }

    // row max: C layout => row M lives in one 16-lane half, VGPR r
#pragma unroll
    for (int r = 0; r < 8; ++r) {
        float m0 = s0[0][r], m1 = s1[0][r];
#pragma unroll
        for (int j = 1; j < 8; ++j) { m0 = fmaxf(m0, s0[j][r]); m1 = fmaxf(m1, s1[j][r]); }
#pragma unroll
        for (int msk = 8; msk >= 1; msk >>= 1) {
            m0 = fmaxf(m0, __shfl_xor(m0, msk, 16));
            m1 = fmaxf(m1, __shfl_xor(m1, msk, 16));
        }
        if (l15 == 0) { wred[(0 * 8 + wave) * 16 + rb + r] = m0; wred[(1 * 8 + wave) * 16 + rb + r] = m1; }
    }
    __syncthreads();
    if (tid < 32) {
        int m = tid >> 4, row = tid & 15;
        float v = wred[(m * 8 + 0) * 16 + row];
        for (int w = 1; w < 8; ++w) v = fmaxf(v, wred[(m * 8 + w) * 16 + row]);
        rowstat[m * 16 + row] = v;
    }
    __syncthreads();

    // exp + row sums
    float rm0[8], rm1[8];
#pragma unroll
    for (int r = 0; r < 8; ++r) { rm0[r] = rowstat[rb + r]; rm1[r] = rowstat[16 + rb + r]; }
#pragma unroll
    for (int r = 0; r < 8; ++r) {
        float t0 = 0.f, t1 = 0.f;
#pragma unroll
        for (int j = 0; j < 8; ++j) {
            s0[j][r] = __expf(s0[j][r] - rm0[r]); t0 += s0[j][r];
            s1[j][r] = __expf(s1[j][r] - rm1[r]); t1 += s1[j][r];
        }
#pragma unroll
        for (int msk = 8; msk >= 1; msk >>= 1) {
            t0 += __shfl_xor(t0, msk, 16);
            t1 += __shfl_xor(t1, msk, 16);
        }
        if (l15 == 0) { wred[(0 * 8 + wave) * 16 + rb + r] = t0; wred[(1 * 8 + wave) * 16 + rb + r] = t1; }
    }
    __syncthreads();
    if (tid < 32) {
        int m = tid >> 4, row = tid & 15;
        float v = 0.f;
        for (int w = 0; w < 8; ++w) v += wred[(m * 8 + w) * 16 + row];
        rowstat[m * 16 + row] = v;
    }
    __syncthreads();

    // combine branches, write f16 probabilities to LDS
    const float lam = lamsh[0];
#pragma unroll
    for (int r = 0; r < 8; ++r) {
        int row = rb + r;
        float i0 = 1.0f / rowstat[row];
        float i1 = 1.0f / rowstat[16 + row];
#pragma unroll
        for (int j = 0; j < 8; ++j) {
            float p = s0[j][r] * i0 + lam * s1[j][r] * i1;
            P[row * 1024 + (wave + j * 8) * 16 + l15] = (f16)p;
        }
    }
    __syncthreads();

    // ---------------- phase 2: O = P @ V ------------------------------
    v8f o0 = {}, o1 = {};
#pragma unroll
    for (int j = 0; j < 4; ++j) {
        int kc = wave + j * 8;  // 32 K-chunks of 32
        v16h a  = load_a16(P + l15 * 1024 + kc * 32 + off0);
        v16h b0 = *(const v16h*)(Vt + l15 * 1024 + kc * 32 + kh2);
        v16h b1 = *(const v16h*)(Vt + (16 + l15) * 1024 + kc * 32 + kh2);
        o0 = __builtin_amdgcn_wmma_f32_16x16x32_f16(false, a, false, b0, (short)0, o0, false, false);
        o1 = __builtin_amdgcn_wmma_f32_16x16x32_f16(false, a, false, b1, (short)0, o1, false, false);
    }
#pragma unroll
    for (int r = 0; r < 8; ++r) {
        int row = rb + r;
        atomicAdd(&O[row * 32 + l15], o0[r]);
        atomicAdd(&O[row * 32 + 16 + l15], o1[r]);
    }
    __syncthreads();

    // RMS norm + (1 - lam_init) scale, write f16 tokens [b][seq][512]
    if (tid < 16) {
        float ss = 0.f;
        for (int d = 0; d < 32; ++d) { float v = O[tid * 32 + d]; ss += v * v; }
        rmssh[tid] = rsqrtf(ss * (1.0f / 32.0f) + EPSF);
    }
    __syncthreads();
    const float oms = lamsh[1];
    for (int i = tid; i < 512; i += 256) {
        int row = i >> 5, d = i & 31;
        float v = O[i] * rmssh[row] * rmsw[d] * oms;
        aout[((size_t)(b * 1024 + qt * 16 + row)) * 512 + h * 32 + d] = (f16)v;
    }
}

// ---------------------------------------------------------------------------
// Pack attention output tokens into padded NCHW f16 for the final conv
// ---------------------------------------------------------------------------
__global__ __launch_bounds__(256) void pack_apad_kernel(const f16* __restrict__ aout,
                                                        f16* __restrict__ apad) {
    int i = blockIdx.x * 256 + threadIdx.x;
    if (i >= 4 * 512 * 34 * 34) return;
    int xx = i % 34; int t = i / 34;
    int yy = t % 34; t /= 34;
    int c  = t & 511;
    int b  = t >> 9;
    f16 v = (f16)0.f;
    if (xx >= 1 && xx <= 32 && yy >= 1 && yy <= 32) {
        v = aout[((size_t)(b * 1024 + (yy - 1) * 32 + (xx - 1))) * 512 + c];
    }
    apad[i] = v;
}

// ---------------------------------------------------------------------------
extern "C" void kernel_launch(void* const* d_in, const int* in_sizes, int n_in,
                              void* d_out, int out_size, void* d_ws, size_t ws_size,
                              hipStream_t stream) {
    const float* x    = (const float*)d_in[0];
    const float* gnw  = (const float*)d_in[1];
    const float* wq   = (const float*)d_in[2];
    const float* wk   = (const float*)d_in[3];
    const float* wv   = (const float*)d_in[4];
    const float* wo   = (const float*)d_in[5];
    const float* lq1  = (const float*)d_in[6];
    const float* lq2  = (const float*)d_in[7];
    const float* lk1  = (const float*)d_in[8];
    const float* lk2  = (const float*)d_in[9];
    const float* li   = (const float*)d_in[10];
    const float* rmsw = (const float*)d_in[11];
    float* out = (float*)d_out;

    char* ws = (char*)d_ws;
    size_t o = 0;
    auto alloc = [&](size_t bytes) { size_t r = o; o += (bytes + 255) & ~(size_t)255; return r; };
    size_t off_stats = alloc(8 * 4);
    size_t off_wq16  = alloc(1024ull * 4608 * 2);
    size_t off_wk16  = alloc(256ull  * 4608 * 2);
    size_t off_wv16  = alloc(128ull  * 4608 * 2);
    size_t off_wo16  = alloc(512ull  * 4608 * 2);
    size_t off_xpad  = alloc(4ull * 512 * 34 * 34 * 2);
    size_t off_tokq  = alloc(4ull * 1024 * 1024 * 2);
    size_t off_tokk  = alloc(4ull * 1024 * 256 * 2);
    size_t off_tokv  = alloc(4ull * 1024 * 128 * 2);
    size_t off_aout  = alloc(4ull * 1024 * 512 * 2);
    size_t off_apad  = alloc(4ull * 512 * 34 * 34 * 2);
    (void)ws_size; (void)in_sizes; (void)n_in; (void)out_size;

    float* stats = (float*)(ws + off_stats);
    f16* wq16 = (f16*)(ws + off_wq16);
    f16* wk16 = (f16*)(ws + off_wk16);
    f16* wv16 = (f16*)(ws + off_wv16);
    f16* wo16 = (f16*)(ws + off_wo16);
    f16* xpad = (f16*)(ws + off_xpad);
    f16* tokq = (f16*)(ws + off_tokq);
    f16* tokk = (f16*)(ws + off_tokk);
    f16* tokv = (f16*)(ws + off_tokv);
    f16* aout = (f16*)(ws + off_aout);
    f16* apad = (f16*)(ws + off_apad);

    // 1) stats + normalized padded f16 activations
    gn_stats_kernel<<<4, 256, 0, stream>>>(x, stats);
    pack_xpad_kernel<<<(4 * 512 * 34 * 34 + 255) / 256, 256, 0, stream>>>(x, gnw, stats, xpad);

    // 2) weights -> f16 (layout already [OC][K])
    f32_to_f16_kernel<<<(4718592 + 255) / 256, 256, 0, stream>>>(wq, wq16, 4718592);
    f32_to_f16_kernel<<<(1179648 + 255) / 256, 256, 0, stream>>>(wk, wk16, 1179648);
    f32_to_f16_kernel<<<(589824  + 255) / 256, 256, 0, stream>>>(wv, wv16, 589824);
    f32_to_f16_kernel<<<(2359296 + 255) / 256, 256, 0, stream>>>(wo, wo16, 2359296);

    // 3) q/k/v convs -> f16 tokens
    conv_wmma_kernel<<<dim3(8, 32, 4), 256, 0, stream>>>(wq16, xpad, tokq, nullptr, nullptr, 1024, 0);
    conv_wmma_kernel<<<dim3(2, 32, 4), 256, 0, stream>>>(wk16, xpad, tokk, nullptr, nullptr, 256, 0);
    conv_wmma_kernel<<<dim3(1, 32, 4), 256, 0, stream>>>(wv16, xpad, tokv, nullptr, nullptr, 128, 0);

    // 4) RoPE in place
    rope_kernel<<<(2097152 + 255) / 256, 256, 0, stream>>>(tokq, 16, 2097152);
    rope_kernel<<<(524288  + 255) / 256, 256, 0, stream>>>(tokk, 4, 524288);

    // 5) differential attention (LDS size allocated unconditionally so host
    //    and device preprocessing cannot diverge on the TDM path)
    diffattn_kernel<<<dim3(64, 16, 4), 256, ATT_SMEM, stream>>>(
        tokq, tokk, tokv, lq1, lq2, lk1, lk2, li, rmsw, aout);

    // 6) output projection conv + residual
    pack_apad_kernel<<<(4 * 512 * 34 * 34 + 255) / 256, 256, 0, stream>>>(aout, apad);
    conv_wmma_kernel<<<dim3(4, 32, 4), 256, 0, stream>>>(wo16, apad, nullptr, out, x, 512, 1);
}
